// TCN_15564961480881
// MI455X (gfx1250) — matrix operand order repaired and checked by reading, not compiled, not run on hardware
//
#include <hip/hip_runtime.h>
#include <math.h>

typedef __attribute__((ext_vector_type(16))) _Float16 v16h;
typedef __attribute__((ext_vector_type(8)))  _Float16 v8h;
typedef __attribute__((ext_vector_type(8)))  float    v8f;

// Per-MLP padded W^T layout (identical for all 4 MLPs used), biases folded in
// as weight row k == Kdim of each layer (augmented-input trick):
//  L0: [80][32]  = 2560 halfs   (in dims 10/7/12 padded to K=32; bias row K0)
//  L1: [80][96]  = 7680 halfs   (bias row 80)
//  L2: [80][96]  = 7680 halfs   (bias row 80)
//  L3: [16][96]  = 1536 halfs   (out dims 4/3/1 padded to N=16; bias row 80)
static constexpr int kWtHalfs = 19456;
static constexpr int kWtBytes = 38912;    // per-MLP region (ws and LDS)
static constexpr int kNFrag   = 38;       // WMMA A-fragments per MLP

// ---------------- device helpers ----------------

// Half-wave swap: every lane gets the value from lane^16 (pure VALU,
// v_permlanex16_b32 with identity lane-selects; no LDS round-trip).
__device__ __forceinline__ int swap16(int v) {
  return __builtin_amdgcn_permlanex16(v, v, 0x76543210, 0xfedcba98, false,
                                      false);
}

__device__ __forceinline__ v8f wmma16(v16h a, v16h b, v8f c) {
  // v_wmma_f32_16x16x32_f16  D = A(16x32) * B(32x16) + C
  return __builtin_amdgcn_wmma_f32_16x16x32_f16(false, a, false, b, (short)0, c,
                                                false, false);
}

// A-fragment (weights) from LDS: lane holds row (16c + lane&15) of W^T,
// halfs [kb..kb+15] with kb = 32s + (laneHi ? 16 : 0).
__device__ __forceinline__ v16h load_wfrag(const _Float16* w, int Kpad, int c,
                                           int s, int lane) {
  const int row = c * 16 + (lane & 15);
  const int kb  = s * 32 + ((lane & 16) ? 16 : 0);
  const _Float16* p = w + row * Kpad + kb;
  v8h lo = *(const v8h*)p;          // ds_load_b128
  v8h hi = *(const v8h*)(p + 8);    // ds_load_b128
  v16h a;
#pragma unroll
  for (int i = 0; i < 8; ++i) { a[i] = lo[i]; a[i + 8] = hi[i]; }
  return a;
}

// Load ALL 38 weight fragments of one MLP into registers (304 VGPRs).
// Fragment index map: L0: c  |  L1: 5+c*3+s  |  L2: 20+c*3+s  |  L3: 35+s.
__device__ __forceinline__ void load_weights_reg(const _Float16* lw, int lane,
                                                 v16h* Wt) {
#pragma unroll
  for (int c = 0; c < 5; ++c) Wt[c] = load_wfrag(lw, 32, c, 0, lane);
#pragma unroll
  for (int c = 0; c < 5; ++c)
#pragma unroll
    for (int s = 0; s < 3; ++s)
      Wt[5 + c * 3 + s] = load_wfrag(lw + 2560, 96, c, s, lane);
#pragma unroll
  for (int c = 0; c < 5; ++c)
#pragma unroll
    for (int s = 0; s < 3; ++s)
      Wt[20 + c * 3 + s] = load_wfrag(lw + 10240, 96, c, s, lane);
#pragma unroll
  for (int s = 0; s < 3; ++s)
    Wt[35 + s] = load_wfrag(lw + 17920, 96, 0, s, lane);
}

// ReLU as a single v_med3_f32 per element: median(x, 0, +inf) == max(x, 0),
// avoiding the canonicalize+max pair fmaxf() lowers to.
__device__ __forceinline__ v8f relu8(v8f a) {
#pragma unroll
  for (int i = 0; i < 8; ++i)
    a[i] = __builtin_amdgcn_fmed3f(a[i], 0.0f, __builtin_inff());
  return a;
}

// Build next-layer B fragment (K-slice 32s..32s+31) from D chunks 2s (dlo)
// and 2s+1 (dhi) using f32->f16 packing plus a half-wave lane swap.
__device__ __forceinline__ v16h make_bfrag(v8f dlo, v8f dhi, bool laneLo) {
  union U { _Float16 h[8]; int i[4]; };
  U P, Q, S, T, BL, BH;
#pragma unroll
  for (int k = 0; k < 8; ++k) {
    P.h[k] = (_Float16)dlo[k];
    Q.h[k] = (_Float16)dhi[k];
  }
#pragma unroll
  for (int k = 0; k < 4; ++k) {
    S.i[k] = swap16(P.i[k]);   // partner's P
    T.i[k] = swap16(Q.i[k]);   // partner's Q
  }
#pragma unroll
  for (int k = 0; k < 4; ++k) {
    BL.i[k] = laneLo ? P.i[k] : T.i[k];    // b[0..7]
    BH.i[k] = laneLo ? S.i[k] : Q.i[k];    // b[8..15]
  }
  v16h b;
#pragma unroll
  for (int k = 0; k < 8; ++k) { b[k] = BL.h[k]; b[k + 8] = BH.h[k]; }
  return b;
}

// Last B fragment of a transition: covers K=64..95, where chunk 5 (dims
// 80..95) is the implicit bias activation: dim80 = 1.0, rest 0.  The constant
// f16 1.0 (0x3C00) is injected directly for the upper half-wave; the spurious
// 1 this also places at K=88 multiplies all-zero weight rows (harmless).
__device__ __forceinline__ v16h make_bfrag_last(v8f dlo, bool laneLo) {
  union U { _Float16 h[8]; int i[4]; };
  U P, S, BL, BH;
#pragma unroll
  for (int k = 0; k < 8; ++k) P.h[k] = (_Float16)dlo[k];
#pragma unroll
  for (int k = 0; k < 4; ++k) S.i[k] = swap16(P.i[k]);
  const int q0 = 0x00003C00;   // halfs (1.0, 0.0)
#pragma unroll
  for (int k = 0; k < 4; ++k) {
    BL.i[k] = laneLo ? P.i[k] : (k == 0 ? q0 : 0);
    BH.i[k] = laneLo ? S.i[k] : (k == 0 ? q0 : 0);
  }
  v16h b;
#pragma unroll
  for (int k = 0; k < 8; ++k) { b[k] = BL.h[k]; b[k + 8] = BH.h[k]; }
  return b;
}

// 4-layer MLP in transposed WMMA form, weights (with folded biases) in
// registers. Output dims live in result[0..3] of lanes 0..15.
__device__ __forceinline__ v8f run_mlp4_reg(const v16h* Wt, v16h b0, int lane) {
  const bool laneLo = (lane & 16) == 0;
  v16h bf[3];
  bf[0] = b0;
  v8f D[5];
  // L0
#pragma unroll
  for (int c = 0; c < 5; ++c) {
    v8f acc = {};
    D[c] = relu8(wmma16(Wt[c], bf[0], acc));
  }
  bf[0] = make_bfrag(D[0], D[1], laneLo);
  bf[1] = make_bfrag(D[2], D[3], laneLo);
  bf[2] = make_bfrag_last(D[4], laneLo);
  // L1
#pragma unroll
  for (int c = 0; c < 5; ++c) {
    v8f acc = {};
#pragma unroll
    for (int s = 0; s < 3; ++s) acc = wmma16(Wt[5 + c * 3 + s], bf[s], acc);
    D[c] = relu8(acc);
  }
  bf[0] = make_bfrag(D[0], D[1], laneLo);
  bf[1] = make_bfrag(D[2], D[3], laneLo);
  bf[2] = make_bfrag_last(D[4], laneLo);
  // L2
#pragma unroll
  for (int c = 0; c < 5; ++c) {
    v8f acc = {};
#pragma unroll
    for (int s = 0; s < 3; ++s) acc = wmma16(Wt[20 + c * 3 + s], bf[s], acc);
    D[c] = relu8(acc);
  }
  bf[0] = make_bfrag(D[0], D[1], laneLo);
  bf[1] = make_bfrag(D[2], D[3], laneLo);
  bf[2] = make_bfrag_last(D[4], laneLo);
  // L3 (linear; bias already folded in)
  v8f acc = {};
#pragma unroll
  for (int s = 0; s < 3; ++s) acc = wmma16(Wt[35 + s], bf[s], acc);
  return acc;
}

// Same MLP but streaming weight fragments from LDS (for the second MLP in
// gnn_edge2, whose fragments don't fit in registers alongside the first).
__device__ __forceinline__ v8f run_mlp4_lds(const _Float16* lw, v16h b0,
                                            int lane) {
  const bool laneLo = (lane & 16) == 0;
  v16h bf[3];
  bf[0] = b0;
  v8f D[5];
#pragma unroll
  for (int c = 0; c < 5; ++c) {
    v8f acc = {};
    D[c] = relu8(wmma16(load_wfrag(lw, 32, c, 0, lane), bf[0], acc));
  }
  bf[0] = make_bfrag(D[0], D[1], laneLo);
  bf[1] = make_bfrag(D[2], D[3], laneLo);
  bf[2] = make_bfrag_last(D[4], laneLo);
#pragma unroll
  for (int c = 0; c < 5; ++c) {
    v8f acc = {};
#pragma unroll
    for (int s = 0; s < 3; ++s)
      acc = wmma16(load_wfrag(lw + 2560, 96, c, s, lane), bf[s], acc);
    D[c] = relu8(acc);
  }
  bf[0] = make_bfrag(D[0], D[1], laneLo);
  bf[1] = make_bfrag(D[2], D[3], laneLo);
  bf[2] = make_bfrag_last(D[4], laneLo);
#pragma unroll
  for (int c = 0; c < 5; ++c) {
    v8f acc = {};
#pragma unroll
    for (int s = 0; s < 3; ++s)
      acc = wmma16(load_wfrag(lw + 10240, 96, c, s, lane), bf[s], acc);
    D[c] = relu8(acc);
  }
  bf[0] = make_bfrag(D[0], D[1], laneLo);
  bf[1] = make_bfrag(D[2], D[3], laneLo);
  bf[2] = make_bfrag_last(D[4], laneLo);
  v8f acc = {};
#pragma unroll
  for (int s = 0; s < 3; ++s)
    acc = wmma16(load_wfrag(lw + 17920, 96, 0, s, lane), bf[s], acc);
  return acc;
}

__device__ __forceinline__ void block_copy_lds(char* dst,
                                               const char* __restrict__ src,
                                               int bytes) {
  int4* d = (int4*)dst;
  const int4* s = (const int4*)src;
  const int n = bytes >> 4;
  for (int i = threadIdx.x; i < n; i += blockDim.x) d[i] = s[i];
  __syncthreads();
}

// ---------------- kernels ----------------

// Transpose + f32->f16 convert one MLP's weights into padded W^T panels,
// folding each layer's bias into weight row k == Kdim.
__global__ void __launch_bounds__(256) gnn_prep_mlp(
    const float* __restrict__ W0, const float* __restrict__ B0,
    const float* __restrict__ W1, const float* __restrict__ B1,
    const float* __restrict__ W2, const float* __restrict__ B2,
    const float* __restrict__ W3, const float* __restrict__ B3,
    int K0, int N3, char* __restrict__ dst) {
  const int t = blockIdx.x * blockDim.x + threadIdx.x;
  _Float16* wt = (_Float16*)dst;
  if (t < kWtHalfs) {
    int off, Kpad, Kdim, Ndim;
    const float *W, *Bv;
    if (t < 2560) {
      off = 0;     Kpad = 32; Kdim = K0; Ndim = 80; W = W0; Bv = B0;
    } else if (t < 10240) {
      off = 2560;  Kpad = 96; Kdim = 80; Ndim = 80; W = W1; Bv = B1;
    } else if (t < 17920) {
      off = 10240; Kpad = 96; Kdim = 80; Ndim = 80; W = W2; Bv = B2;
    } else {
      off = 17920; Kpad = 96; Kdim = 80; Ndim = N3; W = W3; Bv = B3;
    }
    const int local = t - off;
    const int n = local / Kpad;
    const int k = local % Kpad;
    float v = 0.0f;
    if (n < Ndim) {
      if (k < Kdim)       v = W[k * Ndim + n];
      else if (k == Kdim) v = Bv[n];          // folded bias row
    }
    wt[t] = (_Float16)v;
  }
}

__global__ void gnn_zero_f32(float* __restrict__ p, int n) {
  const int i = blockIdx.x * blockDim.x + threadIdx.x;
  if (i < n) p[i] = 0.0f;
}

// Pass 1: e1 = R1_mlp([x[dst], x[src], ea]) per edge; scatter-add into aggr.
__attribute__((amdgpu_waves_per_eu(2)))
__global__ void __launch_bounds__(256) gnn_edge1(
    const float* __restrict__ x, const int* __restrict__ ei,
    const float* __restrict__ ea, const char* __restrict__ wsMlp,
    float* __restrict__ e1, float* __restrict__ aggr, int E) {
  __shared__ __align__(16) char smem[kWtBytes];
  block_copy_lds(smem, wsMlp, kWtBytes);
  const int lane = threadIdx.x & 31;
  const int wave = threadIdx.x >> 5;
  v16h Wt[kNFrag];
  load_weights_reg((const _Float16*)smem, lane, Wt);  // loop-invariant, 304 VGPRs
  const int wstride = gridDim.x * (blockDim.x >> 5);
  const int tiles = (E + 15) >> 4;
  for (int t = blockIdx.x * (blockDim.x >> 5) + wave; t < tiles; t += wstride) {
    const int e = t * 16 + (lane & 15);
    const bool valid = e < E;
    int dN = 0;
    v16h b0 = {};
    if (lane < 16 && valid) {
      const int sN = ei[e];
      dN = ei[E + e];
      b0[0] = (_Float16)x[dN * 3 + 0];
      b0[1] = (_Float16)x[dN * 3 + 1];
      b0[2] = (_Float16)x[dN * 3 + 2];
      b0[3] = (_Float16)x[sN * 3 + 0];
      b0[4] = (_Float16)x[sN * 3 + 1];
      b0[5] = (_Float16)x[sN * 3 + 2];
      b0[6] = (_Float16)ea[e * 4 + 0];
      b0[7] = (_Float16)ea[e * 4 + 1];
      b0[8] = (_Float16)ea[e * 4 + 2];
      b0[9] = (_Float16)ea[e * 4 + 3];
      b0[10] = (_Float16)1.0f;             // bias activation (K0 = 10)
    }
    v8f o = run_mlp4_reg(Wt, b0, lane);
    if (lane < 16 && valid) {
      *(float4*)(e1 + (size_t)e * 4) = make_float4(o[0], o[1], o[2], o[3]);
      atomicAdd(&aggr[dN * 4 + 0], o[0]);
      atomicAdd(&aggr[dN * 4 + 1], o[1]);
      atomicAdd(&aggr[dN * 4 + 2], o[2]);
      atomicAdd(&aggr[dN * 4 + 3], o[3]);
    }
  }
}

// Pass 2: x1 = O1_mlp([x, aggr]) per node.
__attribute__((amdgpu_waves_per_eu(2)))
__global__ void __launch_bounds__(256) gnn_node(
    const float* __restrict__ x, const float* __restrict__ aggr,
    const char* __restrict__ wsMlp, float* __restrict__ x1, int N) {
  __shared__ __align__(16) char smem[kWtBytes];
  block_copy_lds(smem, wsMlp, kWtBytes);
  const int lane = threadIdx.x & 31;
  const int wave = threadIdx.x >> 5;
  v16h Wt[kNFrag];
  load_weights_reg((const _Float16*)smem, lane, Wt);
  const int wstride = gridDim.x * (blockDim.x >> 5);
  const int tiles = (N + 15) >> 4;
  for (int t = blockIdx.x * (blockDim.x >> 5) + wave; t < tiles; t += wstride) {
    const int v = t * 16 + (lane & 15);
    const bool valid = v < N;
    v16h b0 = {};
    if (lane < 16 && valid) {
      b0[0] = (_Float16)x[v * 3 + 0];
      b0[1] = (_Float16)x[v * 3 + 1];
      b0[2] = (_Float16)x[v * 3 + 2];
      b0[3] = (_Float16)aggr[v * 4 + 0];
      b0[4] = (_Float16)aggr[v * 4 + 1];
      b0[5] = (_Float16)aggr[v * 4 + 2];
      b0[6] = (_Float16)aggr[v * 4 + 3];
      b0[7] = (_Float16)1.0f;              // bias activation (K0 = 7)
    }
    v8f o = run_mlp4_reg(Wt, b0, lane);
    if (lane < 16 && valid)
      *(float4*)(x1 + (size_t)v * 4) = make_float4(o[0], o[1], o[2], 0.f);
  }
}

// Pass 3: e2 = R2_mlp([x1[dst], x1[src], e1]); out = sigmoid(W_mlp([ea,e1,e2])).
// R2's fragments live in registers; the W MLP streams fragments from LDS.
__attribute__((amdgpu_waves_per_eu(2)))
__global__ void __launch_bounds__(256) gnn_edge2(
    const float* __restrict__ ea, const int* __restrict__ ei,
    const float* __restrict__ x1, const float* __restrict__ e1,
    const char* __restrict__ wsMlp2, float* __restrict__ out, int E) {
  __shared__ __align__(16) char smem[2 * kWtBytes];
  block_copy_lds(smem, wsMlp2, 2 * kWtBytes);
  const _Float16* lwW = (const _Float16*)(smem + kWtBytes);
  const int lane = threadIdx.x & 31;
  const int wave = threadIdx.x >> 5;
  v16h Wt[kNFrag];
  load_weights_reg((const _Float16*)smem, lane, Wt);   // R2 weights
  const int wstride = gridDim.x * (blockDim.x >> 5);
  const int tiles = (E + 15) >> 4;
  for (int t = blockIdx.x * (blockDim.x >> 5) + wave; t < tiles; t += wstride) {
    const int e = t * 16 + (lane & 15);
    const bool valid = e < E;
    float4 eav = make_float4(0.f, 0.f, 0.f, 0.f);
    float4 e1v = eav;
    v16h bR = {};
    if (lane < 16 && valid) {
      const int sN = ei[e];
      const int dN = ei[E + e];
      const float4 x1d = ((const float4*)x1)[dN];
      const float4 x1s = ((const float4*)x1)[sN];
      e1v = ((const float4*)e1)[e];
      eav = ((const float4*)ea)[e];
      bR[0] = (_Float16)x1d.x; bR[1] = (_Float16)x1d.y; bR[2] = (_Float16)x1d.z;
      bR[3] = (_Float16)x1s.x; bR[4] = (_Float16)x1s.y; bR[5] = (_Float16)x1s.z;
      bR[6] = (_Float16)e1v.x; bR[7] = (_Float16)e1v.y;
      bR[8] = (_Float16)e1v.z; bR[9] = (_Float16)e1v.w;
      bR[10] = (_Float16)1.0f;             // bias activation (K0 = 10)
    }
    v8f o2 = run_mlp4_reg(Wt, bR, lane);
    v16h bW = {};
    if (lane < 16 && valid) {
      bW[0]  = (_Float16)eav.x; bW[1]  = (_Float16)eav.y;
      bW[2]  = (_Float16)eav.z; bW[3]  = (_Float16)eav.w;
      bW[4]  = (_Float16)e1v.x; bW[5]  = (_Float16)e1v.y;
      bW[6]  = (_Float16)e1v.z; bW[7]  = (_Float16)e1v.w;
      bW[8]  = (_Float16)o2[0]; bW[9]  = (_Float16)o2[1];
      bW[10] = (_Float16)o2[2]; bW[11] = (_Float16)o2[3];
      bW[12] = (_Float16)1.0f;             // bias activation (K0 = 12)
    }
    v8f zf = run_mlp4_lds(lwW, bW, lane);
    if (lane < 16 && valid) out[e] = 1.0f / (1.0f + expf(-zf[0]));
  }
}

// ---------------- host launcher ----------------
// d_in order: 0:x 1:edge_index 2:edge_attr, then param leaves:
//   in1_R1: 3..10   in1_O: 11..18   in2_R1: 19..26   in2_O: 27..34 (UNUSED)
//   W:      35..42
extern "C" void kernel_launch(void* const* d_in, const int* in_sizes, int n_in,
                              void* d_out, int out_size, void* d_ws,
                              size_t ws_size, hipStream_t stream) {
  (void)n_in; (void)out_size; (void)ws_size;
  const float* x  = (const float*)d_in[0];
  const int*   ei = (const int*)d_in[1];   // [2,E] (src row, dst row)
  const float* ea = (const float*)d_in[2];
  const int N = in_sizes[0] / 3;
  const int E = in_sizes[2] / 4;

  char* ws = (char*)d_ws;
  char* wsR1 = ws + 0 * (size_t)kWtBytes;
  char* wsO1 = ws + 1 * (size_t)kWtBytes;
  char* wsR2 = ws + 2 * (size_t)kWtBytes;   // R2 and W contiguous for edge2
  char* wsW  = ws + 3 * (size_t)kWtBytes;
  size_t off = 4 * (size_t)kWtBytes;
  float* e1 = (float*)(ws + off);   off += (size_t)E * 4 * sizeof(float);
  float* ag = (float*)(ws + off);   off += (size_t)N * 4 * sizeof(float);
  float* x1 = (float*)(ws + off);

  auto PF = [&](int i) { return (const float*)d_in[i]; };

  const dim3 pb(256), pg((kWtHalfs + 255) / 256);
  gnn_prep_mlp<<<pg, pb, 0, stream>>>(PF(3), PF(4), PF(5), PF(6), PF(7), PF(8),
                                      PF(9), PF(10), 10, 4, wsR1);
  gnn_prep_mlp<<<pg, pb, 0, stream>>>(PF(11), PF(12), PF(13), PF(14), PF(15),
                                      PF(16), PF(17), PF(18), 7, 3, wsO1);
  gnn_prep_mlp<<<pg, pb, 0, stream>>>(PF(19), PF(20), PF(21), PF(22), PF(23),
                                      PF(24), PF(25), PF(26), 10, 4, wsR2);
  gnn_prep_mlp<<<pg, pb, 0, stream>>>(PF(35), PF(36), PF(37), PF(38), PF(39),
                                      PF(40), PF(41), PF(42), 12, 1, wsW);
  gnn_zero_f32<<<(N * 4 + 255) / 256, 256, 0, stream>>>(ag, N * 4);

  gnn_edge1<<<1024, 256, 0, stream>>>(x, ei, ea, wsR1, e1, ag, E);
  gnn_node<<<512, 256, 0, stream>>>(x, ag, wsO1, x1, N);
  gnn_edge2<<<1024, 256, 0, stream>>>(ea, ei, x1, e1, wsR2, (float*)d_out, E);
}
